// IRLGAD_37555194036835
// MI455X (gfx1250) — compile-verified, matrix operand order checked by reading, not run on hardware
//
#include <hip/hip_runtime.h>
#include <hip/hip_bf16.h>
#include <math.h>

#define HIDN 128
#define HEADS 8
#define DPH 16
#define RH 128
#define BETA 0.1f
#define INV_BETA 10.0f
#define NEG_INF (-INFINITY)

typedef __bf16 v16bf __attribute__((ext_vector_type(16)));
typedef float  v8f   __attribute__((ext_vector_type(8)));

// ---------- helpers ----------
__device__ __forceinline__ void atomicMaxF32(float* addr, float val) {
  // order-preserving int trick; location only ever holds -inf or written values
  if (val >= 0.f) atomicMax((int*)addr, __float_as_int(val));
  else            atomicMin((unsigned int*)addr, __float_as_uint(val));
}
__device__ __forceinline__ float finite0(float m) {
  return __builtin_isfinite(m) ? m : 0.f;
}

// ---------- generic fill ----------
__global__ void k_fill(float* p, float v, size_t n) {
  size_t i = (size_t)blockIdx.x * blockDim.x + threadIdx.x;
  if (i < n) p[i] = v;
}

// ---------- weight convert+transpose: Wt[n][k] = (bf16) W[row_off+k][n] ----------
__global__ void k_wt_bf16(const float* __restrict__ W, __bf16* __restrict__ Wt, int row_off) {
  int i = blockIdx.x * blockDim.x + threadIdx.x;   // 0..16383
  int n = i >> 7, k = i & 127;
  Wt[(size_t)n * HIDN + k] = (__bf16)W[(size_t)(row_off + k) * HIDN + n];
}

// ---------- WMMA GEMM: C[N,128] = A[N,128] (f32->bf16) x Wt^T (bf16, n-major) ----------
__global__ __launch_bounds__(256) void k_gemm_wmma(const float* __restrict__ A,
                                                   const __bf16* __restrict__ Wt,
                                                   float* __restrict__ C, int nrows) {
  const int wave = threadIdx.x >> 5;        // 8 waves = 8 col tiles (128 cols)
  const int lane = threadIdx.x & 31;
  const int hi   = lane >> 4;               // half-wave selects K sub-block (+8)
  const int mtile = blockIdx.x;
  int m  = mtile * 16 + (lane & 15);
  int mc = m < nrows ? m : nrows - 1;       // keep EXEC all-ones; guard on store
  const int n = wave * 16 + (lane & 15);
  const float*  arow = A  + (size_t)mc * HIDN;
  const __bf16* brow = Wt + (size_t)n  * HIDN;
  v8f c = {};
#pragma unroll
  for (int kb = 0; kb < HIDN; kb += 32) {
    const int k0 = kb + hi * 8;
    v16bf a, b;
#pragma unroll
    for (int j = 0; j < 8; ++j) {
      // 16-bit A layout: VGPR0-3 -> K=k0..k0+7, VGPR4-7 -> K=k0+16..k0+23
      a[j]     = (__bf16)arow[k0 + j];
      a[j + 8] = (__bf16)arow[k0 + 16 + j];
      b[j]     = brow[k0 + j];
      b[j + 8] = brow[k0 + 16 + j];
    }
    c = __builtin_amdgcn_wmma_f32_16x16x32_bf16(false, a, false, b, (short)0, c,
                                                false, false);
  }
  const int mo = mtile * 16 + hi * 8;
#pragma unroll
  for (int v = 0; v < 8; ++v) {
    int row = mo + v;
    if (row < nrows) C[(size_t)row * HIDN + n] = c[v];
  }
}

// ---------- per-(node,head) attention logits ----------
__global__ void k_logits(const float* __restrict__ H, const float* __restrict__ asrc,
                         const float* __restrict__ adst, float* als, float* ald, int N) {
  int i = blockIdx.x * blockDim.x + threadIdx.x;
  if (i >= N * HEADS) return;
  int nd = i >> 3, h = i & 7;
  const float* hp = H + (size_t)nd * HIDN + h * DPH;
  float s = 0.f, d = 0.f;
#pragma unroll
  for (int j = 0; j < DPH; ++j) {
    s += hp[j] * asrc[h * DPH + j];
    d += hp[j] * adst[h * DPH + j];
  }
  als[i] = s; ald[i] = d;
}

__device__ __forceinline__ float edge_logit(const float* als, const float* ald,
                                            int s, int d, int h) {
  float z = als[s * HEADS + h] + ald[d * HEADS + h];
  return z >= 0.f ? z : 0.2f * z;            // leaky_relu 0.2
}

// ---------- segment softmax over dst (3 passes) ----------
__global__ void k_att_max(const int* src, const int* dst, const float* als,
                          const float* ald, float* Msm, int E_) {
  int e = blockIdx.x * blockDim.x + threadIdx.x;
  if (e >= E_) return;
  int s = src[e], d = dst[e];
#pragma unroll
  for (int h = 0; h < HEADS; ++h)
    atomicMaxF32(&Msm[d * HEADS + h], edge_logit(als, ald, s, d, h));
}
__global__ void k_att_sum(const int* src, const int* dst, const float* als,
                          const float* ald, const float* Msm, float* Ssm, int E_) {
  int e = blockIdx.x * blockDim.x + threadIdx.x;
  if (e >= E_) return;
  int s = src[e], d = dst[e];
#pragma unroll
  for (int h = 0; h < HEADS; ++h) {
    float z = edge_logit(als, ald, s, d, h);
    atomicAdd(&Ssm[d * HEADS + h], expf(z - finite0(Msm[d * HEADS + h])));
  }
}
__global__ void k_att_use(const int* src, const int* dst, const float* als,
                          const float* ald, const float* Msm, const float* Ssm,
                          const float* __restrict__ Hsrc, float* agg, float* amean,
                          int E_, int do_agg) {
  int e = blockIdx.x * blockDim.x + threadIdx.x;
  if (e >= E_) return;
  int s = src[e], d = dst[e];
  float acc = 0.f;
#pragma unroll
  for (int h = 0; h < HEADS; ++h) {
    float m0 = finite0(Msm[d * HEADS + h]);
    float lse = logf(fmaxf(Ssm[d * HEADS + h], 1e-38f)) + m0;
    float a = expf(edge_logit(als, ald, s, d, h) - lse);
    acc += a;
    if (do_agg) {
#pragma unroll
      for (int t = 0; t < DPH; ++t)
        atomicAdd(&agg[(size_t)d * HIDN + h * DPH + t],
                  a * Hsrc[(size_t)s * HIDN + h * DPH + t]);
    }
  }
  amean[e] = acc * (1.f / HEADS);
}

// ---------- elementwise elu (in place) ----------
__global__ void k_elu(float* p, size_t n) {
  size_t i = (size_t)blockIdx.x * blockDim.x + threadIdx.x;
  if (i < n) { float v = p[i]; p[i] = v > 0.f ? v : expf(v) - 1.f; }
}

// ---------- degrees ----------
__global__ void k_deg(const int* src, const int* dst, float* degout, float* degin, int E_) {
  int e = blockIdx.x * blockDim.x + threadIdx.x;
  if (e >= E_) return;
  atomicAdd(&degout[src[e]], 1.f);
  atomicAdd(&degin[dst[e]], 1.f);
}

// ---------- structural reward MLP ----------
__global__ __launch_bounds__(256) void k_rstr(const int* src, const int* dst,
    const float* degout, const float* degin, const float* Ws1, const float* bs1,
    const float* ws2, const float* bs2, float* rstr, int E_) {
  __shared__ float sW[4 * RH], sb[RH], sw2[RH];
  for (int i = threadIdx.x; i < 4 * RH; i += blockDim.x) sW[i] = Ws1[i];
  for (int i = threadIdx.x; i < RH; i += blockDim.x) { sb[i] = bs1[i]; sw2[i] = ws2[i]; }
  __syncthreads();
  int e = blockIdx.x * blockDim.x + threadIdx.x;
  if (e >= E_) return;
  float f0 = degout[src[e]], f1 = degin[dst[e]];
  float f2 = log1pf(f0), f3 = log1pf(f1);
  float acc = 0.f;
#pragma unroll 4
  for (int j = 0; j < RH; ++j) {
    float z = sb[j] + f0 * sW[j] + f1 * sW[RH + j] + f2 * sW[2 * RH + j] + f3 * sW[3 * RH + j];
    acc += fmaxf(z, 0.f) * sw2[j];
  }
  rstr[e] = acc + bs2[0];
}

// ---------- semantic reward per edge (one wave32 per edge) ----------
__global__ __launch_bounds__(256) void k_rsem(const int* src, const int* dst,
    const float* __restrict__ U, const float* __restrict__ Vf, const float* bm1,
    const float* wm2, const float* bm2, const float* lam1, const float* rstr,
    float* rout, int E_) {
  int wid  = (blockIdx.x * blockDim.x + threadIdx.x) >> 5;
  int lane = threadIdx.x & 31;
  if (wid >= E_) return;
  const float* up = U  + (size_t)src[wid] * HIDN;
  const float* vp = Vf + (size_t)dst[wid] * HIDN;
  float acc = 0.f;
#pragma unroll
  for (int j = lane; j < HIDN; j += 32) {
    float z = up[j] + vp[j] + bm1[j];
    acc += fmaxf(z, 0.f) * wm2[j];
  }
#pragma unroll
  for (int o = 16; o; o >>= 1) acc += __shfl_xor(acc, o, 32);
  if (lane == 0) rout[wid] = rstr[wid] + lam1[0] * (acc + bm2[0]);
}

// ---------- SVI (segment LSE over src) ----------
__global__ void k_svi_max(const int* src, const int* dst, const float* r,
                          const float* V, float* Mv, int E_) {
  int e = blockIdx.x * blockDim.x + threadIdx.x;
  if (e >= E_) return;
  atomicMaxF32(&Mv[src[e]], (r[e] + V[dst[e]]) * INV_BETA);
}
__global__ void k_svi_sum(const int* src, const int* dst, const float* r,
                          const float* V, const float* Mv, float* Sv, int E_) {
  int e = blockIdx.x * blockDim.x + threadIdx.x;
  if (e >= E_) return;
  float z = (r[e] + V[dst[e]]) * INV_BETA;
  atomicAdd(&Sv[src[e]], expf(z - finite0(Mv[src[e]])));
}
__global__ void k_svi_node(const float* Mv, const float* Sv, const float* degout,
                           float* V, int N) {
  int i = blockIdx.x * blockDim.x + threadIdx.x;
  if (i >= N) return;
  float m0 = finite0(Mv[i]);
  float s = Sv[i];
  float lse = s > 0.f ? logf(fmaxf(s, 1e-38f)) + m0 : NEG_INF;
  V[i] = degout[i] > 0.f ? BETA * lse : 0.f;
}
__global__ void k_logz(const float* Mv, const float* Sv, float* logz, int N) {
  int i = blockIdx.x * blockDim.x + threadIdx.x;
  if (i >= N) return;
  float m0 = finite0(Mv[i]);
  float s = Sv[i];
  logz[i] = s > 0.f ? logf(fmaxf(s, 1e-38f)) + m0 : NEG_INF;
}
__global__ void k_svi_logpi(const int* src, const int* dst, const float* r,
                            const float* V, const float* logz, float* lps, int E_) {
  int e = blockIdx.x * blockDim.x + threadIdx.x;
  if (e >= E_) return;
  lps[e] = (r[e] + V[dst[e]]) * INV_BETA - logz[src[e]];
}

// ---------- observed policy log-prob (segment LSE over src) ----------
__global__ void k_obs_max(const int* src, const float* a, float* Mv, int E_) {
  int e = blockIdx.x * blockDim.x + threadIdx.x;
  if (e >= E_) return;
  atomicMaxF32(&Mv[src[e]], logf(a[e] + 1e-12f));
}
__global__ void k_obs_sum(const int* src, const float* a, const float* Mv,
                          float* Sv, int E_) {
  int e = blockIdx.x * blockDim.x + threadIdx.x;
  if (e >= E_) return;
  float la = logf(a[e] + 1e-12f);
  atomicAdd(&Sv[src[e]], expf(la - finite0(Mv[src[e]])));
}
__global__ void k_obs_logpi(const int* src, const float* a, const float* Mv,
                            const float* Sv, float* lpo, int E_) {
  int e = blockIdx.x * blockDim.x + threadIdx.x;
  if (e >= E_) return;
  int s = src[e];
  float la = logf(a[e] + 1e-12f);
  lpo[e] = la - (logf(fmaxf(Sv[s], 1e-38f)) + finite0(Mv[s]));
}

// ---------- NLL / KL accumulation ----------
__global__ void k_accum(const int* src, const float* lpo, const float* lps,
                        float* nll, float* score, int E_) {
  int e = blockIdx.x * blockDim.x + threadIdx.x;
  if (e >= E_) return;
  float p = expf(lpo[e]);
  atomicAdd(&nll[src[e]], -p * lps[e]);
  atomicAdd(&score[src[e]], p * (lpo[e] - lps[e]));
}

// ---------- mean reduction for loss ----------
__global__ __launch_bounds__(256) void k_loss(const float* nll, float* out, int N) {
  __shared__ float sm[256];
  float acc = 0.f;
  for (size_t i = (size_t)blockIdx.x * blockDim.x + threadIdx.x; i < (size_t)N;
       i += (size_t)gridDim.x * blockDim.x)
    acc += nll[i];
  sm[threadIdx.x] = acc;
  __syncthreads();
  for (int o = 128; o; o >>= 1) {
    if (threadIdx.x < (unsigned)o) sm[threadIdx.x] += sm[threadIdx.x + o];
    __syncthreads();
  }
  if (threadIdx.x == 0) atomicAdd(out, sm[0] / (float)N);
}

extern "C" void kernel_launch(void* const* d_in, const int* in_sizes, int n_in,
                              void* d_out, int out_size, void* d_ws, size_t ws_size,
                              hipStream_t stream) {
  (void)n_in; (void)out_size; (void)ws_size;
  const float* x   = (const float*)d_in[0];
  const int*   ei  = (const int*)d_in[1];
  const float* W0  = (const float*)d_in[2];
  const float* as0 = (const float*)d_in[3];
  const float* ad0 = (const float*)d_in[4];
  const float* W1  = (const float*)d_in[5];
  const float* as1 = (const float*)d_in[6];
  const float* ad1 = (const float*)d_in[7];
  const float* Ws1 = (const float*)d_in[8];
  const float* bs1 = (const float*)d_in[9];
  const float* ws2 = (const float*)d_in[10];
  const float* bs2 = (const float*)d_in[11];
  const float* Wm1 = (const float*)d_in[12];
  const float* bm1 = (const float*)d_in[13];
  const float* wm2 = (const float*)d_in[14];
  const float* bm2 = (const float*)d_in[15];
  const float* lam = (const float*)d_in[16];

  const int N = in_sizes[0] / HIDN;
  const int E = in_sizes[1] / 2;
  const int* src = ei;
  const int* dst = ei + E;

  // workspace layout (floats, then bf16 weight tiles)
  const size_t NHf = (size_t)N * HIDN;
  float* B1   = (float*)d_ws;          // H0, later U
  float* B2   = B1 + NHf;              // agg -> h1
  float* B3   = B2 + NHf;              // H1, later Vfeat
  float* als  = B3 + NHf;              // [N*8]
  float* ald  = als + (size_t)N * HEADS;
  float* Msm  = ald + (size_t)N * HEADS;
  float* Ssm  = Msm + (size_t)N * HEADS;
  float* a1   = Ssm + (size_t)N * HEADS; // [E]
  float* a2   = a1 + (size_t)E;
  float* rstr = a2 + (size_t)E;
  float* redg = rstr + (size_t)E;
  float* lps  = redg + (size_t)E;
  float* lpo  = lps + (size_t)E;
  float* dgo  = lpo + (size_t)E;       // [N]
  float* dgi  = dgo + N;
  float* Vv   = dgi + N;
  float* Mv   = Vv + N;
  float* Sv   = Mv + N;
  float* lgz  = Sv + N;
  float* nll  = lgz + N;
  __bf16* WT0 = (__bf16*)(nll + N);
  __bf16* WT1 = WT0 + 16384;
  __bf16* WTt = WT1 + 16384;
  __bf16* WTb = WTt + 16384;

  const int TB = 256;
  const int gE  = (E + TB - 1) / TB;
  const int gN  = (N + TB - 1) / TB;
  const int gNH = (N * HEADS + TB - 1) / TB;
  const int gEw = (E + 7) / 8;         // wave32 per edge
  const int gMM = (N + 15) / 16;
  auto fill = [&](float* p, float v, size_t n) {
    k_fill<<<(unsigned)((n + TB - 1) / TB), TB, 0, stream>>>(p, v, n);
  };

  // weights -> bf16 transposed
  k_wt_bf16<<<64, TB, 0, stream>>>(W0, WT0, 0);
  k_wt_bf16<<<64, TB, 0, stream>>>(W1, WT1, 0);
  k_wt_bf16<<<64, TB, 0, stream>>>(Wm1, WTt, 0);    // top half of Wm1
  k_wt_bf16<<<64, TB, 0, stream>>>(Wm1, WTb, HIDN); // bottom half

  // ---- GAT layer 1 ----
  k_gemm_wmma<<<gMM, TB, 0, stream>>>(x, WT0, B1, N);               // H0
  k_logits<<<gNH, TB, 0, stream>>>(B1, as0, ad0, als, ald, N);
  fill(Msm, NEG_INF, (size_t)N * HEADS);
  fill(Ssm, 0.f, (size_t)N * HEADS);
  k_att_max<<<gE, TB, 0, stream>>>(src, dst, als, ald, Msm, E);
  k_att_sum<<<gE, TB, 0, stream>>>(src, dst, als, ald, Msm, Ssm, E);
  fill(B2, 0.f, NHf);
  k_att_use<<<gE, TB, 0, stream>>>(src, dst, als, ald, Msm, Ssm, B1, B2, a1, E, 1);
  k_elu<<<(unsigned)((NHf + TB - 1) / TB), TB, 0, stream>>>(B2, NHf); // h1

  // ---- GAT layer 2 (only attention needed) ----
  k_gemm_wmma<<<gMM, TB, 0, stream>>>(B2, WT1, B3, N);              // H1
  k_logits<<<gNH, TB, 0, stream>>>(B3, as1, ad1, als, ald, N);
  fill(Msm, NEG_INF, (size_t)N * HEADS);
  fill(Ssm, 0.f, (size_t)N * HEADS);
  k_att_max<<<gE, TB, 0, stream>>>(src, dst, als, ald, Msm, E);
  k_att_sum<<<gE, TB, 0, stream>>>(src, dst, als, ald, Msm, Ssm, E);
  k_att_use<<<gE, TB, 0, stream>>>(src, dst, als, ald, Msm, Ssm, B3, Vv, a2, E, 0);

  // ---- degrees + structural reward ----
  fill(dgo, 0.f, N); fill(dgi, 0.f, N);
  k_deg<<<gE, TB, 0, stream>>>(src, dst, dgo, dgi, E);
  k_rstr<<<gE, TB, 0, stream>>>(src, dst, dgo, dgi, Ws1, bs1, ws2, bs2, rstr, E);

  // ---- outputs ----
  fill((float*)d_out, 0.f, (size_t)N + 1);
  fill(nll, 0.f, N);

  for (int t = 0; t < 2; ++t) {
    const float* hid = (t == 0) ? x : B2;
    k_gemm_wmma<<<gMM, TB, 0, stream>>>(hid, WTt, B1, N);           // U
    k_gemm_wmma<<<gMM, TB, 0, stream>>>(hid, WTb, B3, N);           // Vfeat
    k_rsem<<<gEw, TB, 0, stream>>>(src, dst, B1, B3, bm1, wm2, bm2, lam, rstr, redg, E);

    const float* at = (t == 0) ? a1 : a2;
    fill(Mv, NEG_INF, N); fill(Sv, 0.f, N);
    k_obs_max<<<gE, TB, 0, stream>>>(src, at, Mv, E);
    k_obs_sum<<<gE, TB, 0, stream>>>(src, at, Mv, Sv, E);
    k_obs_logpi<<<gE, TB, 0, stream>>>(src, at, Mv, Sv, lpo, E);

    fill(Vv, 0.f, N);
    for (int it = 0; it < 5; ++it) {
      fill(Mv, NEG_INF, N); fill(Sv, 0.f, N);
      k_svi_max<<<gE, TB, 0, stream>>>(src, dst, redg, Vv, Mv, E);
      k_svi_sum<<<gE, TB, 0, stream>>>(src, dst, redg, Vv, Mv, Sv, E);
      k_svi_node<<<gN, TB, 0, stream>>>(Mv, Sv, dgo, Vv, N);
    }
    fill(Mv, NEG_INF, N); fill(Sv, 0.f, N);
    k_svi_max<<<gE, TB, 0, stream>>>(src, dst, redg, Vv, Mv, E);
    k_svi_sum<<<gE, TB, 0, stream>>>(src, dst, redg, Vv, Mv, Sv, E);
    k_logz<<<gN, TB, 0, stream>>>(Mv, Sv, lgz, N);
    k_svi_logpi<<<gE, TB, 0, stream>>>(src, dst, redg, Vv, lgz, lps, E);

    k_accum<<<gE, TB, 0, stream>>>(src, lpo, lps, nll, (float*)d_out + 1, E);
  }
  k_loss<<<256, TB, 0, stream>>>(nll, (float*)d_out, N);
}
